// MultiHeadedAttention_23691039604948
// MI455X (gfx1250) — matrix operand order repaired and checked by reading, not compiled, not run on hardware
//
#include <hip/hip_runtime.h>
#include <hip/hip_bf16.h>

typedef __attribute__((ext_vector_type(16))) _Float16 v16h;
typedef __attribute__((ext_vector_type(8)))  _Float16 v8h;
typedef __attribute__((ext_vector_type(8)))  float    v8f;
typedef __attribute__((ext_vector_type(4)))  float    v4f;

#define ATTN_B  4
#define ATTN_S  2048
#define ATTN_D  1024
#define ATTN_H  16
#define ATTN_HD 64

// 1/sqrt(HD) * log2(e): softmax computed in base-2 domain (v_exp_f32 native);
// folded into the Q fragment once, in f16.
#define ATTN_SCALE 0.1803368801111204f

static __device__ __forceinline__ v16h combine16(v8h lo, v8h hi) {
  v16h r;
#pragma unroll
  for (int j = 0; j < 8; ++j) { r[j] = lo[j]; r[8 + j] = hi[j]; }
  return r;
}

// ---------------------------------------------------------------------------
// CDNA5 async global->LDS copy (ASYNCcnt-tracked, no VGPR round trip).
// INST_OFFSET applies to both LDS and global addresses (ISA 08 §4.4).
// ---------------------------------------------------------------------------
static __device__ __forceinline__ void async_ld_b128(const void* lds_dst,
                                                     const void* gsrc,
                                                     int ioff) {
  uint32_t lds_off = (uint32_t)(uintptr_t)lds_dst;  // low 32 bits = LDS offset
  uint64_t gaddr = (uint64_t)(uintptr_t)gsrc;
  asm volatile("global_load_async_to_lds_b128 %0, %1, off offset:%2"
               :: "v"(lds_off), "v"(gaddr), "i"(ioff)
               : "memory");
}
static __device__ __forceinline__ void async_wait0() {
  asm volatile("s_wait_asynccnt 0" ::: "memory");
}

// ---------------------------------------------------------------------------
// DPP16 cross-lane reductions within 16-lane rows (no LDS-pipe bpermute).
// ---------------------------------------------------------------------------
template <int CTRL>
static __device__ __forceinline__ float dpp_movf(float x) {
  return __int_as_float(__builtin_amdgcn_update_dpp(
      0, __float_as_int(x), CTRL, 0xF, 0xF, true));
}
static __device__ __forceinline__ float row_max16(float x) {
  x = fmaxf(x, dpp_movf<0xB1>(x));   // quad_perm(1,0,3,2)  : xor 1
  x = fmaxf(x, dpp_movf<0x4E>(x));   // quad_perm(2,3,0,1)  : xor 2
  x = fmaxf(x, dpp_movf<0x141>(x));  // row_half_mirror     : pair quads
  x = fmaxf(x, dpp_movf<0x140>(x));  // row_mirror          : pair octets
  return x;
}
static __device__ __forceinline__ float row_sum16(float x) {
  x += dpp_movf<0xB1>(x);
  x += dpp_movf<0x4E>(x);
  x += dpp_movf<0x141>(x);
  x += dpp_movf<0x140>(x);
  return x;
}

// ---------------------------------------------------------------------------
// f32 -> f16 elementwise conversion
// ---------------------------------------------------------------------------
__global__ void cvt_f32_to_f16(const float* __restrict__ s,
                               _Float16* __restrict__ d, int n) {
  int i = (blockIdx.x * blockDim.x + threadIdx.x) * 4;
  if (i + 3 < n) {
    v4f f = *(const v4f*)(s + i);
#pragma unroll
    for (int j = 0; j < 4; ++j) d[i + j] = (_Float16)f[j];
  }
}

// ---------------------------------------------------------------------------
// Tiled f16 WMMA GEMM:  out[M,N] = A[M,K] * Bw[K,N] + bias[N]
// Block tile 128x128, 8 waves (2x4), wave tile 64x32, K-step 32.
// Double-buffered: next A tile via async DMA + next B tile global->VGPR are
// issued before the current tile's WMMAs; one barrier per K-step.
// ---------------------------------------------------------------------------
template <bool OUT_F16>
__global__ __launch_bounds__(256)
void gemm_f16_wmma(const _Float16* __restrict__ A,
                   const _Float16* __restrict__ Bw,
                   const float* __restrict__ bias,
                   void* __restrict__ outp, int M, int N, int K) {
  __shared__ __align__(32) _Float16 lA[2][128 * 32];   // [m][k]
  __shared__ __align__(32) _Float16 lBt[2][128 * 32];  // [n][k] (transposed)

  const int tid  = threadIdx.x;
  const int lane = tid & 31;
  const int w    = tid >> 5;
  const int mw   = w & 1;
  const int nw   = w >> 1;
  const int M0   = blockIdx.x * 128;
  const int N0   = blockIdx.y * 128;
  const int lx   = lane & 15;
  const int hsel = lane >> 4;
  const int klo  = hsel ? 8 : 0;
  const int kk0  = hsel ? 16 : 0;

  v8f acc[4][2];
  const v8f vzero = {};
#pragma unroll
  for (int mt = 0; mt < 4; ++mt)
#pragma unroll
    for (int nt = 0; nt < 2; ++nt) acc[mt][nt] = vzero;

  const int arow = tid >> 1, ahf = tid & 1;
  const int krow = tid >> 3, nch = tid & 7;
  const _Float16* agsrc = A + (size_t)(M0 + arow) * K + ahf * 16;

  // Prologue: stage tile 0
  {
    async_ld_b128(&lA[0][arow * 32 + ahf * 16], agsrc, 0);
    async_ld_b128(&lA[0][arow * 32 + ahf * 16], agsrc, 16);
    const _Float16* wsrc = Bw + (size_t)krow * N + N0 + nch * 16;
    v8h b0 = *(const v8h*)wsrc;
    v8h b1 = *(const v8h*)(wsrc + 8);
#pragma unroll
    for (int j = 0; j < 8; ++j) {
      lBt[0][(nch * 16 + j) * 32 + krow]     = b0[j];
      lBt[0][(nch * 16 + 8 + j) * 32 + krow] = b1[j];
    }
    async_wait0();
    __syncthreads();
  }

  const int NKC = K / 32;
  for (int kc = 0; kc < NKC; ++kc) {
    const int cur = kc & 1;
    const int nxt = cur ^ 1;
    const bool hn = (kc + 1) < NKC;
    v8h nb0 = {}, nb1 = {};
    if (hn) {  // issue next tile's loads before computing (block-uniform)
      const int kn = (kc + 1) * 32;
      async_ld_b128(&lA[nxt][arow * 32 + ahf * 16], agsrc + kn, 0);
      async_ld_b128(&lA[nxt][arow * 32 + ahf * 16], agsrc + kn, 16);
      const _Float16* wsrc = Bw + (size_t)(kn + krow) * N + N0 + nch * 16;
      nb0 = *(const v8h*)wsrc;
      nb1 = *(const v8h*)(wsrc + 8);
    }

    v16h afr[4];
#pragma unroll
    for (int mt = 0; mt < 4; ++mt) {
      const _Float16* p = &lA[cur][(mw * 64 + mt * 16 + lx) * 32];
      afr[mt] = combine16(*(const v8h*)(p + kk0), *(const v8h*)(p + kk0 + 8));
    }
    v16h bfr[2];
#pragma unroll
    for (int nt = 0; nt < 2; ++nt)
      bfr[nt] = *(const v16h*)&lBt[cur][(nw * 32 + nt * 16 + lx) * 32 + kk0];

#pragma unroll
    for (int mt = 0; mt < 4; ++mt)
#pragma unroll
      for (int nt = 0; nt < 2; ++nt)
        acc[mt][nt] = __builtin_amdgcn_wmma_f32_16x16x32_f16(
            false, afr[mt], false, bfr[nt], (short)0, acc[mt][nt], false, false);

    if (hn) {  // scatter next B tile (buffer last read before prior barrier)
#pragma unroll
      for (int j = 0; j < 8; ++j) {
        lBt[nxt][(nch * 16 + j) * 32 + krow]     = nb0[j];
        lBt[nxt][(nch * 16 + 8 + j) * 32 + krow] = nb1[j];
      }
    }
    async_wait0();
    __syncthreads();
  }

#pragma unroll
  for (int nt = 0; nt < 2; ++nt) {
    const int n = N0 + nw * 32 + nt * 16 + lx;
    const float bn = bias[n];
#pragma unroll
    for (int mt = 0; mt < 4; ++mt) {
#pragma unroll
      for (int i = 0; i < 8; ++i) {
        const int m = M0 + mw * 64 + mt * 16 + i + 8 * hsel;
        const float v = acc[mt][nt][i] + bn;
        if (OUT_F16)
          ((_Float16*)outp)[(size_t)m * N + n] = (_Float16)v;
        else
          ((float*)outp)[(size_t)m * N + n] = v;
      }
    }
  }
}

// ---------------------------------------------------------------------------
// Flash attention: block = (b, h, 128-query tile); 8 waves x 16 queries.
// 64-key tiles (16 WMMAs per softmax pass), double-buffered K (async DMA)
// and V (VGPR transpose scatter), DPP16 reductions, exp2-domain softmax,
// scale pre-folded into Q.
// ---------------------------------------------------------------------------
__global__ __launch_bounds__(256)
void flash_attn_f16_wmma(const _Float16* __restrict__ Q,
                         const _Float16* __restrict__ Kk,
                         const _Float16* __restrict__ V,
                         _Float16* __restrict__ O) {
  __shared__ __align__(32) _Float16 ldsK[2][64 * 64];   // [key][dim]  2x8 KB
  __shared__ __align__(32) _Float16 ldsVt[2][64 * 64];  // [dim][key]  2x8 KB
  __shared__ __align__(32) _Float16 ldsP[8 * 16 * 64];  // wave P scratch 16 KB

  const int tid  = threadIdx.x;
  const int lane = tid & 31;
  const int w    = tid >> 5;
  const int lx   = lane & 15;
  const int hsel = lane >> 4;
  const int klo  = hsel ? 8 : 0;
  const int kk0  = hsel ? 16 : 0;

  const int nqb = ATTN_S / 128;
  const int qb  = blockIdx.x % nqb;
  const int h   = (blockIdx.x / nqb) % ATTN_H;
  const int b   = blockIdx.x / (nqb * ATTN_H);
  const int q0  = qb * 128 + w * 16;

  // Q A-fragments (16 rows x 64 head-dims), softmax scale folded in (f16)
  const _Float16* qrow =
      Q + ((size_t)(b * ATTN_S + q0 + lx)) * ATTN_D + h * ATTN_HD;
  const _Float16 qs = (_Float16)ATTN_SCALE;
  v16h aq[2];
#pragma unroll
  for (int c = 0; c < 2; ++c) {
    aq[c] = combine16(*(const v8h*)(qrow + c * 32 + kk0),
                      *(const v8h*)(qrow + c * 32 + kk0 + 8));
#pragma unroll
    for (int j = 0; j < 16; ++j) aq[c][j] *= qs;
  }

  v8f o[4];
  const v8f vzero = {};
#pragma unroll
  for (int d = 0; d < 4; ++d) o[d] = vzero;
  float mrow[8], lrow[8];
#pragma unroll
  for (int i = 0; i < 8; ++i) { mrow[i] = -3.0e38f; lrow[i] = 0.0f; }

  const int row4 = tid >> 2;  // key row 0..63
  const int ch4  = tid & 3;   // dim chunk of 16
  const size_t srcBase =
      ((size_t)(b * ATTN_S + row4)) * ATTN_D + h * ATTN_HD + ch4 * 16;
  const size_t srcStep = (size_t)64 * ATTN_D;  // 64 keys

  // Prologue: stage key tile 0
  {
    async_ld_b128(&ldsK[0][row4 * 64 + ch4 * 16], Kk + srcBase, 0);
    async_ld_b128(&ldsK[0][row4 * 64 + ch4 * 16], Kk + srcBase, 16);
    v8h vv0 = *(const v8h*)(V + srcBase);
    v8h vv1 = *(const v8h*)(V + srcBase + 8);
#pragma unroll
    for (int j = 0; j < 8; ++j) {
      ldsVt[0][(ch4 * 16 + j) * 64 + row4]     = vv0[j];
      ldsVt[0][(ch4 * 16 + 8 + j) * 64 + row4] = vv1[j];
    }
    async_wait0();
    __syncthreads();
  }

  const int NIT = ATTN_S / 64;  // 32
  for (int it = 0; it < NIT; ++it) {
    const int cur = it & 1;
    const int nxt = cur ^ 1;
    const bool hn = (it + 1) < NIT;
    v8h nv0 = {}, nv1 = {};
    if (hn) {  // issue next tile's loads before computing (block-uniform)
      const size_t sn = srcBase + (size_t)(it + 1) * srcStep;
      async_ld_b128(&ldsK[nxt][row4 * 64 + ch4 * 16], Kk + sn, 0);
      async_ld_b128(&ldsK[nxt][row4 * 64 + ch4 * 16], Kk + sn, 16);
      nv0 = *(const v8h*)(V + sn);
      nv1 = *(const v8h*)(V + sn + 8);
    }

    // scores S = (Q*scale) * K^T : four 16x16 key tiles, 2 WMMAs each
    v8f sc[4];
#pragma unroll
    for (int t = 0; t < 4; ++t) {
      v8f s = vzero;
#pragma unroll
      for (int c = 0; c < 2; ++c) {
        v16h bk =
            *(const v16h*)&ldsK[cur][(t * 16 + lx) * 64 + c * 32 + kk0];
        s = __builtin_amdgcn_wmma_f32_16x16x32_f16(false, aq[c], false, bk,
                                                   (short)0, s, false, false);
      }
      sc[t] = s;
    }

    // Online softmax in base-2 domain
    float p[4][8], alpha[8];
#pragma unroll
    for (int i = 0; i < 8; ++i) {
      float vmax = fmaxf(fmaxf(sc[0][i], sc[1][i]), fmaxf(sc[2][i], sc[3][i]));
      vmax = row_max16(vmax);
      const float mnew = fmaxf(mrow[i], vmax);
      alpha[i] = exp2f(mrow[i] - mnew);
      mrow[i] = mnew;
      float ps = 0.0f;
#pragma unroll
      for (int t = 0; t < 4; ++t) {
        p[t][i] = exp2f(sc[t][i] - mnew);
        ps += p[t][i];
      }
      ps = row_sum16(ps);
      lrow[i] = lrow[i] * alpha[i] + ps;
    }
#pragma unroll
    for (int d = 0; d < 4; ++d)
#pragma unroll
      for (int i = 0; i < 8; ++i) o[d][i] *= alpha[i];

    // P: C-layout -> A-layout via per-wave LDS bounce (DS in-order per wave)
    _Float16* pw = &ldsP[w * 1024];
#pragma unroll
    for (int t = 0; t < 4; ++t)
#pragma unroll
      for (int i = 0; i < 8; ++i)
        pw[(i + 8 * hsel) * 64 + t * 16 + lx] = (_Float16)p[t][i];
    __builtin_amdgcn_wave_barrier();
    const _Float16* pr = pw + lx * 64;
    v16h ap[2];
#pragma unroll
    for (int c = 0; c < 2; ++c)
      ap[c] = combine16(*(const v8h*)(pr + c * 32 + kk0),
                        *(const v8h*)(pr + c * 32 + kk0 + 8));

    // O += P * V : 4 dim tiles x 2 key chunks = 8 WMMAs
#pragma unroll
    for (int d = 0; d < 4; ++d) {
#pragma unroll
      for (int c = 0; c < 2; ++c) {
        v16h bvf =
            *(const v16h*)&ldsVt[cur][(d * 16 + lx) * 64 + c * 32 + kk0];
        o[d] = __builtin_amdgcn_wmma_f32_16x16x32_f16(false, ap[c], false, bvf,
                                                      (short)0, o[d], false,
                                                      false);
      }
    }

    if (hn) {  // scatter next V tile (buffer last read before prior barrier)
#pragma unroll
      for (int j = 0; j < 8; ++j) {
        ldsVt[nxt][(ch4 * 16 + j) * 64 + row4]     = nv0[j];
        ldsVt[nxt][(ch4 * 16 + 8 + j) * 64 + row4] = nv1[j];
      }
    }
    async_wait0();
    __syncthreads();
  }

  float invl[8];
#pragma unroll
  for (int i = 0; i < 8; ++i) invl[i] = 1.0f / lrow[i];
#pragma unroll
  for (int d = 0; d < 4; ++d) {
#pragma unroll
    for (int i = 0; i < 8; ++i) {
      const int r = q0 + i + 8 * hsel;
      const int cc = h * ATTN_HD + d * 16 + lx;
      O[((size_t)(b * ATTN_S + r)) * ATTN_D + cc] =
          (_Float16)(o[d][i] * invl[i]);
    }
  }
}

// ---------------------------------------------------------------------------
extern "C" void kernel_launch(void* const* d_in, const int* in_sizes, int n_in,
                              void* d_out, int out_size, void* d_ws,
                              size_t ws_size, hipStream_t stream) {
  (void)in_sizes; (void)n_in; (void)out_size; (void)ws_size;
  const float* xq = (const float*)d_in[0];
  const float* xk = (const float*)d_in[1];
  const float* xv = (const float*)d_in[2];
  const float* wq = (const float*)d_in[3];
  const float* bq = (const float*)d_in[4];
  const float* wk = (const float*)d_in[5];
  const float* bk = (const float*)d_in[6];
  const float* wv = (const float*)d_in[7];
  const float* bv = (const float*)d_in[8];
  const float* wo = (const float*)d_in[9];
  const float* bo = (const float*)d_in[10];
  float* out = (float*)d_out;

  const size_t nBSD = (size_t)ATTN_B * ATTN_S * ATTN_D;
  const size_t nDD  = (size_t)ATTN_D * ATTN_D;

  _Float16* xq16 = (_Float16*)d_ws;
  _Float16* xk16 = xq16 + nBSD;
  _Float16* xv16 = xk16 + nBSD;
  _Float16* q16  = xv16 + nBSD;
  _Float16* k16  = q16 + nBSD;
  _Float16* v16  = k16 + nBSD;
  _Float16* at16 = v16 + nBSD;
  _Float16* wq16 = at16 + nBSD;
  _Float16* wk16 = wq16 + nDD;
  _Float16* wv16 = wk16 + nDD;
  _Float16* wo16 = wv16 + nDD;

  const int CT = 256;
  const int gBSD = (int)((nBSD / 4 + CT - 1) / CT);
  const int gDD  = (int)((nDD / 4 + CT - 1) / CT);
  cvt_f32_to_f16<<<gBSD, CT, 0, stream>>>(xq, xq16, (int)nBSD);
  cvt_f32_to_f16<<<gBSD, CT, 0, stream>>>(xk, xk16, (int)nBSD);
  cvt_f32_to_f16<<<gBSD, CT, 0, stream>>>(xv, xv16, (int)nBSD);
  cvt_f32_to_f16<<<gDD, CT, 0, stream>>>(wq, wq16, (int)nDD);
  cvt_f32_to_f16<<<gDD, CT, 0, stream>>>(wk, wk16, (int)nDD);
  cvt_f32_to_f16<<<gDD, CT, 0, stream>>>(wv, wv16, (int)nDD);
  cvt_f32_to_f16<<<gDD, CT, 0, stream>>>(wo, wo16, (int)nDD);

  const int M = ATTN_B * ATTN_S;
  const int N = ATTN_D, K = ATTN_D;
  dim3 gg(M / 128, N / 128);
  gemm_f16_wmma<true><<<gg, 256, 0, stream>>>(xq16, wq16, bq, q16, M, N, K);
  gemm_f16_wmma<true><<<gg, 256, 0, stream>>>(xk16, wk16, bk, k16, M, N, K);
  gemm_f16_wmma<true><<<gg, 256, 0, stream>>>(xv16, wv16, bv, v16, M, N, K);

  flash_attn_f16_wmma<<<ATTN_B * ATTN_H * (ATTN_S / 128), 256, 0, stream>>>(
      q16, k16, v16, at16);

  gemm_f16_wmma<false><<<gg, 256, 0, stream>>>(at16, wo16, bo, out, M, N, K);
}
